// dyRouting2_74912819577420
// MI455X (gfx1250) — matrix operand order repaired and checked by reading, not compiled, not run on hardware
//
#include <hip/hip_runtime.h>

// ---------------------------------------------------------------------------
// MI455X (gfx1250) transformer block, compute-bound on bf16 WMMA GEMMs
// (~120 GFLOP vs ~100MB HBM => memory floor ~5us). All matrix operands are
// kept in WMMA-fragment-swizzled global layouts so every fragment load is a
// pair of contiguous global_load_b128 (no packing VALU, almost no LDS), and
// the 192MB L2 provides operand reuse. GEMM inner loop: 12 b128 loads + 8
// v_wmma_f32_16x16x32_bf16, no barriers.
// ---------------------------------------------------------------------------

#define H_DIM 1024
#define NHEAD 16
#define DHEAD 64
#define I_DIM 4096
#define BATCH 4
#define SEQ   1024
#define MROWS (BATCH * SEQ) // 4096

typedef __attribute__((ext_vector_type(16))) __bf16 v16bf;
typedef __attribute__((ext_vector_type(8)))  float  v8f;

union FragU { v16bf f; uint4 u[2]; };

__device__ inline unsigned short f2bf(float f) {
  unsigned int u = __float_as_uint(f);
  u += 0x7FFFu + ((u >> 16) & 1u); // round-to-nearest-even
  return (unsigned short)(u >> 16);
}

// --- fragment-swizzled layouts (tile = 512 elems = 1KB, 32B per lane) -------
// A-operand (16M x 32K tile): lane = half*16 + row; elem j: K = half*16+j.
// Tiles ordered row-tile-major, then K-tile.
__device__ inline size_t aswz(int m, int kk, int Kdim) {
  const int mt = m >> 4, r = m & 15;
  const int kt = kk >> 5, din = kk & 31;
  const int half = din >> 4, j = din & 15;
  return (((size_t)mt * (Kdim >> 5) + kt) << 9) + (size_t)((half * 16 + r) * 16 + j);
}
// B-operand (32K x 16N tile): lane = half*16 + col; elem e: K = half*16 + e.
__device__ inline size_t bswz(int kk, int n, int Ndim) {
  const int kt = kk >> 5, half = (kk >> 4) & 1, e = kk & 15;
  const int nt = n >> 4, c = n & 15;
  return (((size_t)kt * (Ndim >> 4) + nt) << 9) + (size_t)((half * 16 + c) * 16 + e);
}
// k stored as B-operand of Q@K^T (fragment K-dim = head dim d, N = key)
__device__ inline size_t kswz(int b, int h, int key, int d) {
  const size_t tile = ((((size_t)b * NHEAD + h) * (SEQ / 16) + (key >> 4)) * 2) + (d >> 5);
  const int kd = d & 31, half = kd >> 4, e = kd & 15;
  return (tile << 9) + (size_t)((half * 16 + (key & 15)) * 16 + e);
}
// v stored as B-operand of P@V (fragment K-dim = key, N = head dim d)
__device__ inline size_t vswz(int b, int h, int key, int d) {
  const size_t tile = ((((size_t)b * NHEAD + h) * (SEQ / 32) + (key >> 5)) * 4) + (d >> 4);
  const int kd = key & 31, half = kd >> 4, e = kd & 15;
  return (tile << 9) + (size_t)((half * 16 + (d & 15)) * 16 + e);
}

__device__ inline float block_reduce_sum(float v, float* sh) {
#pragma unroll
  for (int m = 16; m >= 1; m >>= 1) v += __shfl_xor(v, m, 32);
  const int lane = threadIdx.x & 31;
  const int w = threadIdx.x >> 5;
  if (lane == 0) sh[w] = v;
  __syncthreads();
  if (w == 0) {
    float tv = (lane < 8) ? sh[lane] : 0.f;
#pragma unroll
    for (int m = 4; m >= 1; m >>= 1) tv += __shfl_xor(tv, m, 32);
    if (lane == 0) sh[0] = tv;
  }
  __syncthreads();
  float r = sh[0];
  __syncthreads();
  return r;
}

// ---------------------------------------------------------------------------
// weights: f32 row-major [K,N] -> bf16 B-fragment-swizzled
// ---------------------------------------------------------------------------
__global__ __launch_bounds__(256) void cvt_w_bswz(
    const float* __restrict__ in, unsigned short* __restrict__ out, int K, int N) {
  int i = blockIdx.x * 256 + threadIdx.x;
  if (i < K * N) {
    const int kk = i / N, n = i - kk * N;
    out[bswz(kk, n, N)] = f2bf(in[i]);
  }
}

// ---------------------------------------------------------------------------
// Dynamic routing: one block per (b,s) row; row lives in registers.
// xs/ys emitted bf16 in A-fragment swizzle (GEMM inputs) + xs f32 (residual).
// ---------------------------------------------------------------------------
__global__ __launch_bounds__(256) void routing_kernel(
    const float* __restrict__ x, const float* __restrict__ y,
    const int* __restrict__ tptr, float* __restrict__ xs_f,
    unsigned short* __restrict__ xs_b, unsigned short* __restrict__ ys_b) {
  __shared__ float sh[8];
  const int row = blockIdx.x;
  const int tid = threadIdx.x;
  const size_t base = (size_t)row * H_DIM + tid * 4;
  float4 xv = *(const float4*)(x + base);
  float4 yv = *(const float4*)(y + base);
  float c1 = 0.5f, c2 = 0.5f, b1 = 0.f, b2 = 0.f;
  const int T = *tptr;
  for (int it = 0; it < T; ++it) {
    if (it > 0) {
      float mx = fmaxf(b1, b2);
      float e1 = expf(b1 - mx), e2 = expf(b2 - mx);
      float inv = 1.f / (e1 + e2);
      c1 = e1 * inv; c2 = e2 * inv;
    }
    float v0 = c1 * xv.x + c2 * yv.x, v1 = c1 * xv.y + c2 * yv.y;
    float v2 = c1 * xv.z + c2 * yv.z, v3 = c1 * xv.w + c2 * yv.w;
    float n2 = block_reduce_sum(v0 * v0 + v1 * v1 + v2 * v2 + v3 * v3, sh);
    float n = sqrtf(n2);
    float s = n / (1.f + n2);
    float d1 = block_reduce_sum(s * (v0 * xv.x + v1 * xv.y + v2 * xv.z + v3 * xv.w), sh);
    float d2 = block_reduce_sum(s * (v0 * yv.x + v1 * yv.y + v2 * yv.z + v3 * yv.w), sh);
    b1 = (it == 0) ? d1 : b1 + d1;
    b2 = (it == 0) ? d2 : b2 + d2;
  }
  float ox[4] = {c1 * xv.x, c1 * xv.y, c1 * xv.z, c1 * xv.w};
  float oy[4] = {c2 * yv.x, c2 * yv.y, c2 * yv.z, c2 * yv.w};
  *(float4*)(xs_f + base) = make_float4(ox[0], ox[1], ox[2], ox[3]);
#pragma unroll
  for (int c = 0; c < 4; ++c) {
    const size_t sw = aswz(row, tid * 4 + c, H_DIM);
    xs_b[sw] = f2bf(ox[c]);
    ys_b[sw] = f2bf(oy[c]);
  }
}

// ---------------------------------------------------------------------------
// bf16 WMMA GEMM, LDS-free: operands pre-swizzled, fragments = 2x b128 loads.
// 128x128 block tile, 8 waves (2x4), each wave 64x32 = 4x2 WMMA accumulators.
// cb_mode: 0=none, 1=A-swizzle(next K = N), 2=kswz, 3=vswz for bf16 output.
// ---------------------------------------------------------------------------
__global__ __launch_bounds__(256) void gemm_bf16_wmma(
    const unsigned short* __restrict__ A, const unsigned short* __restrict__ B,
    const float* __restrict__ bias, float* __restrict__ Cf,
    unsigned short* __restrict__ Cb, int M, int N, int K,
    int fuse_gelu, int cb_mode) {
  const int tid = threadIdx.x;
  const int lane = tid & 31;
  const int half = (lane >> 4) & 1;
  const int l16 = lane & 15;
  const int w = tid >> 5;
  const int wm = w >> 2;  // 0..1
  const int wn = w & 3;   // 0..3
  const int m0 = blockIdx.y * 128;
  const int n0 = blockIdx.x * 128;

  v8f acc[4][2];
#pragma unroll
  for (int i = 0; i < 4; ++i)
#pragma unroll
    for (int j = 0; j < 2; ++j)
#pragma unroll
      for (int e = 0; e < 8; ++e) acc[i][j][e] = 0.f;

  const int Kt = K >> 5, Nt = N >> 4;
  for (int kt = 0; kt < Kt; ++kt) {
    FragU aF[4], bF[2];
#pragma unroll
    for (int i = 0; i < 4; ++i) {
      const int mt = (m0 + wm * 64 + i * 16) >> 4;
      const unsigned short* p = A + (((size_t)mt * Kt + kt) << 9) + lane * 16;
      aF[i].u[0] = ((const uint4*)p)[0];
      aF[i].u[1] = ((const uint4*)p)[1];
    }
#pragma unroll
    for (int j = 0; j < 2; ++j) {
      const int nt = (n0 + wn * 32 + j * 16) >> 4;
      const unsigned short* p = B + (((size_t)kt * Nt + nt) << 9) + lane * 16;
      bF[j].u[0] = ((const uint4*)p)[0];
      bF[j].u[1] = ((const uint4*)p)[1];
    }
    if (kt + 1 < Kt) { // warm L2/L0 for next K tile (global_prefetch_b8)
      const int mt0 = (m0 + wm * 64) >> 4;
      const int nt0 = (n0 + wn * 32) >> 4;
      __builtin_prefetch(A + (((size_t)mt0 * Kt + kt + 1) << 9) + lane * 16, 0, 1);
      __builtin_prefetch(B + (((size_t)(kt + 1) * Nt + nt0) << 9) + lane * 16, 0, 1);
    }
#pragma unroll
    for (int j = 0; j < 2; ++j)
#pragma unroll
      for (int i = 0; i < 4; ++i)
        acc[i][j] = __builtin_amdgcn_wmma_f32_16x16x32_bf16(
            false, aF[i].f, false, bF[j].f, (short)0, acc[i][j], false, false);
  }

#pragma unroll
  for (int j = 0; j < 2; ++j) {
    const int gc = n0 + wn * 32 + j * 16 + l16;
    const float bv = bias ? bias[gc] : 0.f;
#pragma unroll
    for (int i = 0; i < 4; ++i) {
      const int rbase = m0 + wm * 64 + i * 16 + half * 8;
#pragma unroll
      for (int e = 0; e < 8; ++e) {
        const int grow = rbase + e;
        float vv = acc[i][j][e] + bv;
        if (fuse_gelu) vv = 0.5f * vv * (1.f + erff(vv * 0.70710678118f));
        if (Cf) Cf[(size_t)grow * N + gc] = vv;
        if (Cb) {
          const unsigned short hb = f2bf(vv);
          if (cb_mode == 1) {
            Cb[aswz(grow, gc, N)] = hb;
          } else if (cb_mode == 2) {
            Cb[kswz(grow >> 10, gc >> 6, grow & (SEQ - 1), gc & 63)] = hb;
          } else if (cb_mode == 3) {
            Cb[vswz(grow >> 10, gc >> 6, grow & (SEQ - 1), gc & 63)] = hb;
          }
        }
      }
    }
  }
}

// ---------------------------------------------------------------------------
// Flash attention: 4 waves/block, each wave a 16-query tile of one (b,h).
// q/k/v pre-swizzled -> all fragments are 2x global_load_b128 (L2-resident
// reuse), only the probs C->A relayout bounces through a per-wave LDS slab.
// No block barriers at all. ctx written A-swizzled for the Wo GEMM.
// ---------------------------------------------------------------------------
#define LDPS 40

__global__ __launch_bounds__(128) void attention_kernel(
    const unsigned short* __restrict__ q, const unsigned short* __restrict__ k,
    const unsigned short* __restrict__ v, const float* __restrict__ mask,
    unsigned short* __restrict__ ctx) {
  __shared__ unsigned short Ps[4 * 16 * LDPS];

  const int tid = threadIdx.x;
  const int lane = tid & 31;
  const int half = (lane >> 4) & 1;
  const int l16 = lane & 15;
  const int wv = tid >> 5;

  const int qt = blockIdx.x % (SEQ / 64);
  const int bh = blockIdx.x / (SEQ / 64);
  const int b = bh / NHEAD;
  const int h = bh % NHEAD;
  const int q0 = qt * 64 + wv * 16;

  FragU aq[2];
  {
    const int mt = (b * SEQ + q0) >> 4;
#pragma unroll
    for (int h2 = 0; h2 < 2; ++h2) {
      const unsigned short* p =
          q + (((size_t)mt * (H_DIM >> 5) + (h * 2 + h2)) << 9) + lane * 16;
      aq[h2].u[0] = ((const uint4*)p)[0];
      aq[h2].u[1] = ((const uint4*)p)[1];
    }
  }

  float m_s[8], l_s[8];
  v8f o[4];
#pragma unroll
  for (int e = 0; e < 8; ++e) { m_s[e] = -1e30f; l_s[e] = 0.f; }
#pragma unroll
  for (int dt = 0; dt < 4; ++dt)
#pragma unroll
    for (int e = 0; e < 8; ++e) o[dt][e] = 0.f;

  const size_t kbh = ((size_t)b * NHEAD + h) * (SEQ / 16);
  const size_t vbh = ((size_t)b * NHEAD + h) * (SEQ / 32);

  for (int key0 = 0; key0 < SEQ; key0 += 32) {
    // scores: two 16x16 key tiles, each K-dim 64 = 2 WMMAs
    v8f sc[2];
#pragma unroll
    for (int kh = 0; kh < 2; ++kh) {
#pragma unroll
      for (int e = 0; e < 8; ++e) sc[kh][e] = 0.f;
#pragma unroll
      for (int h2 = 0; h2 < 2; ++h2) {
        FragU bK;
        const unsigned short* p =
            k + (((kbh + (key0 >> 4) + kh) * 2 + h2) << 9) + lane * 16;
        bK.u[0] = ((const uint4*)p)[0];
        bK.u[1] = ((const uint4*)p)[1];
        sc[kh] = __builtin_amdgcn_wmma_f32_16x16x32_bf16(
            false, aq[h2].f, false, bK.f, (short)0, sc[kh], false, false);
      }
      const float mk = mask[(size_t)b * SEQ + key0 + kh * 16 + l16];
#pragma unroll
      for (int e = 0; e < 8; ++e) sc[kh][e] = sc[kh][e] * 0.125f + mk;
    }

    // online softmax (row r = e + 8*half spread across 16 lanes)
    float p0[8], p1[8];
#pragma unroll
    for (int e = 0; e < 8; ++e) {
      float cm = fmaxf(sc[0][e], sc[1][e]);
#pragma unroll
      for (int d = 8; d >= 1; d >>= 1) cm = fmaxf(cm, __shfl_xor(cm, d, 32));
      float mn = fmaxf(m_s[e], cm);
      float alpha = expf(m_s[e] - mn);
      p0[e] = expf(sc[0][e] - mn);
      p1[e] = expf(sc[1][e] - mn);
      float rs = p0[e] + p1[e];
#pragma unroll
      for (int d = 8; d >= 1; d >>= 1) rs += __shfl_xor(rs, d, 32);
      l_s[e] = l_s[e] * alpha + rs;
      m_s[e] = mn;
#pragma unroll
      for (int dt = 0; dt < 4; ++dt) o[dt][e] *= alpha;
    }

    // probs: C layout -> per-wave LDS slab -> A-fragment (2x ds_load_b128)
#pragma unroll
    for (int e = 0; e < 8; ++e) {
      const int r = e + half * 8;
      Ps[(wv * 16 + r) * LDPS + l16] = f2bf(p0[e]);
      Ps[(wv * 16 + r) * LDPS + 16 + l16] = f2bf(p1[e]);
    }
    FragU aP;
    aP.u[0] = *(const uint4*)&Ps[(wv * 16 + l16) * LDPS + half * 16];
    aP.u[1] = *(const uint4*)&Ps[(wv * 16 + l16) * LDPS + half * 16 + 8];
#pragma unroll
    for (int dt = 0; dt < 4; ++dt) {
      FragU bV;
      const unsigned short* p =
          v + (((vbh + (key0 >> 5)) * 4 + dt) << 9) + lane * 16;
      bV.u[0] = ((const uint4*)p)[0];
      bV.u[1] = ((const uint4*)p)[1];
      o[dt] = __builtin_amdgcn_wmma_f32_16x16x32_bf16(
          false, aP.f, false, bV.f, (short)0, o[dt], false, false);
    }
  }

#pragma unroll
  for (int dt = 0; dt < 4; ++dt)
#pragma unroll
    for (int e = 0; e < 8; ++e) {
      const float val = o[dt][e] / l_s[e];
      ctx[aswz(b * SEQ + q0 + e + half * 8, h * 64 + dt * 16 + l16, H_DIM)] = f2bf(val);
    }
}

// ---------------------------------------------------------------------------
// RMSNorm with fused residual: out = (in+resid)*rsqrt(mean^2+eps)*g.
// Optional bf16 output in A-fragment swizzle (feeds next GEMM).
// ---------------------------------------------------------------------------
__global__ __launch_bounds__(256) void rmsnorm_kernel(
    const float* __restrict__ in, const float* __restrict__ resid,
    const float* __restrict__ g, float* __restrict__ out_f,
    unsigned short* __restrict__ out_b) {
  __shared__ float sh[8];
  const int row = blockIdx.x;
  const int tid = threadIdx.x;
  const size_t base = (size_t)row * H_DIM + tid * 4;
  float4 vv = *(const float4*)(in + base);
  float4 r4 = *(const float4*)(resid + base);
  vv.x += r4.x; vv.y += r4.y; vv.z += r4.z; vv.w += r4.w;
  float ss = block_reduce_sum(vv.x * vv.x + vv.y * vv.y + vv.z * vv.z + vv.w * vv.w, sh);
  float rs = rsqrtf(ss * (1.f / H_DIM) + 1e-6f);
  const float4 gv = *(const float4*)(g + tid * 4);
  float o0 = vv.x * rs * gv.x, o1 = vv.y * rs * gv.y;
  float o2 = vv.z * rs * gv.z, o3 = vv.w * rs * gv.w;
  if (out_f) *(float4*)(out_f + base) = make_float4(o0, o1, o2, o3);
  if (out_b) {
    float oo[4] = {o0, o1, o2, o3};
#pragma unroll
    for (int c = 0; c < 4; ++c) out_b[aswz(row, tid * 4 + c, H_DIM)] = f2bf(oo[c]);
  }
}

// ---------------------------------------------------------------------------
extern "C" void kernel_launch(void* const* d_in, const int* in_sizes, int n_in,
                              void* d_out, int out_size, void* d_ws, size_t ws_size,
                              hipStream_t stream) {
  (void)in_sizes; (void)n_in; (void)out_size; (void)ws_size;
  const float* x   = (const float*)d_in[0];
  const float* y   = (const float*)d_in[1];
  const float* msk = (const float*)d_in[2];
  const float* Wq  = (const float*)d_in[3];  const float* bq  = (const float*)d_in[4];
  const float* Wk  = (const float*)d_in[5];  const float* bk  = (const float*)d_in[6];
  const float* Wv  = (const float*)d_in[7];  const float* bv  = (const float*)d_in[8];
  const float* Wo  = (const float*)d_in[9];  const float* bo  = (const float*)d_in[10];
  const float* g1  = (const float*)d_in[11];
  const float* Wi  = (const float*)d_in[12]; const float* bi  = (const float*)d_in[13];
  const float* Wo2 = (const float*)d_in[14]; const float* bo2 = (const float*)d_in[15];
  const float* g2  = (const float*)d_in[16];
  const int*   t   = (const int*)d_in[17];

  char* ws = (char*)d_ws;
  size_t off = 0;
  auto carve = [&](size_t bytes) -> void* {
    void* p = ws + off;
    off += (bytes + 255) & ~(size_t)255;
    return p;
  };
  const size_t HH = (size_t)H_DIM * H_DIM, HI = (size_t)H_DIM * I_DIM;
  unsigned short* Wq_b  = (unsigned short*)carve(HH * 2);
  unsigned short* Wk_b  = (unsigned short*)carve(HH * 2);
  unsigned short* Wv_b  = (unsigned short*)carve(HH * 2);
  unsigned short* Wo_b  = (unsigned short*)carve(HH * 2);
  unsigned short* Wi_b  = (unsigned short*)carve(HI * 2);
  unsigned short* Wo2_b = (unsigned short*)carve(HI * 2);
  float*          xs_f  = (float*)carve((size_t)MROWS * H_DIM * 4);
  unsigned short* xs_b  = (unsigned short*)carve((size_t)MROWS * H_DIM * 2);
  unsigned short* ys_b  = (unsigned short*)carve((size_t)MROWS * H_DIM * 2);
  unsigned short* q_b   = (unsigned short*)carve((size_t)MROWS * H_DIM * 2);
  unsigned short* k_b   = (unsigned short*)carve((size_t)MROWS * H_DIM * 2);
  unsigned short* v_b   = (unsigned short*)carve((size_t)MROWS * H_DIM * 2);
  unsigned short* ctx_b = (unsigned short*)carve((size_t)MROWS * H_DIM * 2);
  float*          t0_f  = (float*)carve((size_t)MROWS * H_DIM * 4);
  float*          att_f = (float*)carve((size_t)MROWS * H_DIM * 4);
  unsigned short* att_b = (unsigned short*)carve((size_t)MROWS * H_DIM * 2);
  unsigned short* int_b = (unsigned short*)carve((size_t)MROWS * I_DIM * 2);
  float*          t1_f  = (float*)carve((size_t)MROWS * H_DIM * 4);

  // 1) weights -> bf16 in B-fragment swizzle
  cvt_w_bswz<<<(int)((HH + 255) / 256), 256, 0, stream>>>(Wq, Wq_b, H_DIM, H_DIM);
  cvt_w_bswz<<<(int)((HH + 255) / 256), 256, 0, stream>>>(Wk, Wk_b, H_DIM, H_DIM);
  cvt_w_bswz<<<(int)((HH + 255) / 256), 256, 0, stream>>>(Wv, Wv_b, H_DIM, H_DIM);
  cvt_w_bswz<<<(int)((HH + 255) / 256), 256, 0, stream>>>(Wo, Wo_b, H_DIM, H_DIM);
  cvt_w_bswz<<<(int)((HI + 255) / 256), 256, 0, stream>>>(Wi, Wi_b, H_DIM, I_DIM);
  cvt_w_bswz<<<(int)((HI + 255) / 256), 256, 0, stream>>>(Wo2, Wo2_b, I_DIM, H_DIM);

  // 2) routing -> xs (f32 + A-swz bf16), ys (A-swz bf16)
  routing_kernel<<<MROWS, 256, 0, stream>>>(x, y, t, xs_f, xs_b, ys_b);

  // 3) QKV projections (outputs pre-swizzled for attention)
  dim3 gHH(H_DIM / 128, MROWS / 128);
  gemm_bf16_wmma<<<gHH, 256, 0, stream>>>(xs_b, Wq_b, bq, nullptr, q_b, MROWS, H_DIM, H_DIM, 0, 1);
  gemm_bf16_wmma<<<gHH, 256, 0, stream>>>(ys_b, Wk_b, bk, nullptr, k_b, MROWS, H_DIM, H_DIM, 0, 2);
  gemm_bf16_wmma<<<gHH, 256, 0, stream>>>(ys_b, Wv_b, bv, nullptr, v_b, MROWS, H_DIM, H_DIM, 0, 3);

  // 4) attention -> ctx (A-swz bf16)
  attention_kernel<<<BATCH * NHEAD * (SEQ / 64), 128, 0, stream>>>(q_b, k_b, v_b, msk, ctx_b);

  // 5) output projection + residual RMSNorm
  gemm_bf16_wmma<<<gHH, 256, 0, stream>>>(ctx_b, Wo_b, bo, t0_f, nullptr, MROWS, H_DIM, H_DIM, 0, 0);
  rmsnorm_kernel<<<MROWS, 256, 0, stream>>>(t0_f, xs_f, g1, att_f, att_b);

  // 6) MLP: up (GELU fused, A-swz bf16 out), down, final RMSNorm -> d_out
  dim3 gHI(I_DIM / 128, MROWS / 128);
  gemm_bf16_wmma<<<gHI, 256, 0, stream>>>(att_b, Wi_b, bi, nullptr, int_b, MROWS, I_DIM, H_DIM, 1, 1);
  gemm_bf16_wmma<<<gHH, 256, 0, stream>>>(int_b, Wo2_b, bo2, t1_f, nullptr, MROWS, H_DIM, I_DIM, 0, 0);
  rmsnorm_kernel<<<MROWS, 256, 0, stream>>>(t1_f, att_f, g2, (float*)d_out, nullptr);
}
